// DRINetBlock_2207613190561
// MI455X (gfx1250) — matrix operand order repaired and checked by reading, not compiled, not run on hardware
//
#include <hip/hip_runtime.h>
#include <hip/hip_bf16.h>

// ---------------------------------------------------------------------------
// DRINet-style sparse block on MI455X (gfx1250, wave32, WMMA).
// - All 64x64 GEMMs / 27-tap submanifold conv on v_wmma_f32_16x16x32_bf16.
// - Conv: 27 packed weight matrices (216KB) staged in LDS via
//   global_load_async_to_lds_b128 (ASYNCcnt), B fragments read as ds_load_b128;
//   A gathers software-pipelined one tap ahead; 32 rows/wave (16 WMMA/tap).
//   __launch_bounds__(256,1): LDS already limits WGP to one workgroup, so the
//   VGPR cap is lifted to keep all 8 accumulators + double-buffered A in regs.
// ---------------------------------------------------------------------------

#define M_PTS   120000
#define CCH     64
#define NCLS    20
#define TAPS    27

typedef __attribute__((ext_vector_type(16))) __bf16 v16bf;
typedef __attribute__((ext_vector_type(8)))  float  v8f;

struct __attribute__((aligned(16))) U4 { unsigned x, y, z, w; };

union Frag {
    v16bf    v;
    U4       q[2];
    unsigned u[8];
};

static __device__ __forceinline__ unsigned short f2bf(float f) {
    union { float f; unsigned u; } x; x.f = f;
    unsigned r = x.u + 0x7FFFu + ((x.u >> 16) & 1u);   // round-to-nearest-even
    return (unsigned short)(r >> 16);
}
static __device__ __forceinline__ float bf2f(unsigned short h) {
    union { unsigned u; float f; } x; x.u = ((unsigned)h) << 16;
    return x.f;
}

static __device__ __forceinline__ v8f wmma_bf16(v16bf a, v16bf b, v8f c) {
    return __builtin_amdgcn_wmma_f32_16x16x32_bf16(
        /*neg_a=*/false, a, /*neg_b=*/false, b,
        /*c_mod=*/(short)0, c, /*reuse_a=*/false, /*reuse_b=*/false);
}

// A fragment: 16x32 bf16 tile from a 64-channel bf16 activation row.
// lanes 0-15 (hs=0): K = kb+0..7 , kb+16..23 ; lanes 16-31 (hs=1): +8.
static __device__ __forceinline__ v16bf load_a(const unsigned short* __restrict__ row,
                                               int kb, int hs) {
    Frag f;
    f.q[0] = *reinterpret_cast<const U4*>(row + kb + hs * 8);
    f.q[1] = *reinterpret_cast<const U4*>(row + kb + 16 + hs * 8);
    return f.v;
}
// B fragment from pre-swizzled global weights: 8 consecutive u32 per lane.
static __device__ __forceinline__ v16bf load_b(const unsigned* __restrict__ pk,
                                               int kc, int nt, int ln) {
    Frag f;
    const U4* p = reinterpret_cast<const U4*>(pk + (((kc * 4 + nt) * 32 + ln) * 8));
    f.q[0] = p[0];
    f.q[1] = p[1];
    return f.v;
}

// ---------------------------------------------------------------------------
// Weight pre-swizzle: fp32 [nmat][64(K-in)][64(N-out)] -> bf16 B-fragment
// layout [nmat][kc(2)][nt(4)][lane(32)][vgpr(8)] packed pairs (u32).
// ---------------------------------------------------------------------------
__global__ void pack_w_kernel(const float* __restrict__ W, unsigned* __restrict__ dst,
                              int nmat) {
    int t = blockIdx.x * blockDim.x + threadIdx.x;
    int total = nmat * 2048;
    if (t >= total) return;
    int mat = t / 2048, r = t % 2048;
    int kc = r >> 10, rem = r & 1023;
    int nt = rem >> 8;
    int ln = (rem >> 3) & 31;
    int j  = rem & 7;
    int n  = nt * 16 + (ln & 15);
    int k0 = kc * 32 + (ln >> 4) * 8 + 2 * j + (j >= 4 ? 8 : 0);
    const float* src = W + (size_t)mat * 4096;
    unsigned lo = f2bf(src[(size_t)k0 * 64 + n]);
    unsigned hi = f2bf(src[(size_t)(k0 + 1) * 64 + n]);
    dst[(size_t)mat * 2048 + r] = lo | (hi << 16);
}

__global__ void cvt_bf16_kernel(const float* __restrict__ x,
                                unsigned short* __restrict__ y, int n) {
    int t = blockIdx.x * blockDim.x + threadIdx.x;
    if (t < n) y[t] = f2bf(x[t]);
}

// ---------------------------------------------------------------------------
// GEMM: Y[M,64] = act((Xbf @ W + bpre) * scale + bpost)
// MODE 0: LeakyReLU(0.01) -> bf16 out
// MODE 1: sigmoid epilogue, fused[r,c] (+)= sig * bf2f(ms[r,c])  (f32 out)
// ---------------------------------------------------------------------------
template <int MODE>
__global__ void gemm64_kernel(const unsigned short* __restrict__ Xb,
                              const unsigned* __restrict__ Wpk,
                              const float* __restrict__ bpre,
                              const float* __restrict__ scale,
                              const float* __restrict__ bpost,
                              unsigned short* __restrict__ outb,
                              const unsigned short* __restrict__ msb,
                              float* __restrict__ fused,
                              int accum, int mrows) {
    int wave = threadIdx.x >> 5, ln = threadIdx.x & 31;
    int rowBase = blockIdx.x * 128 + wave * 16;
    if (rowBase >= mrows) return;
    int lr = ln & 15, hs = ln >> 4;
    const unsigned short* arow = Xb + (size_t)(rowBase + lr) * CCH;

    v16bf a0 = load_a(arow, 0, hs);
    v16bf a1 = load_a(arow, 32, hs);
    v16bf b[8];
#pragma unroll
    for (int kc = 0; kc < 2; ++kc)
#pragma unroll
        for (int nt = 0; nt < 4; ++nt)
            b[kc * 4 + nt] = load_b(Wpk, kc, nt, ln);

    v8f acc[4];
#pragma unroll
    for (int nt = 0; nt < 4; ++nt) acc[nt] = (v8f)0.0f;
#pragma unroll
    for (int nt = 0; nt < 4; ++nt) acc[nt] = wmma_bf16(a0, b[nt], acc[nt]);
#pragma unroll
    for (int nt = 0; nt < 4; ++nt) acc[nt] = wmma_bf16(a1, b[4 + nt], acc[nt]);

#pragma unroll
    for (int nt = 0; nt < 4; ++nt) {
        int col = nt * 16 + lr;
        float bp = bpre ? bpre[col] : 0.f;
        float sc = scale[col], bb = bpost[col];
#pragma unroll
        for (int j = 0; j < 8; ++j) {
            int row = rowBase + hs * 8 + j;
            float y = (acc[nt][j] + bp) * sc + bb;
            size_t o = (size_t)row * CCH + col;
            if (MODE == 0) {
                y = y > 0.f ? y : 0.01f * y;
                outb[o] = f2bf(y);
            } else {
                float sig = 1.f / (1.f + __expf(-y));
                float add = sig * bf2f(msb[o]);
                float prev = accum ? fused[o] : 0.f;
                fused[o] = prev + add;
            }
        }
    }
}

// ---------------------------------------------------------------------------
// 27-tap submanifold conv. LDS-staged weights, pipelined gathers,
// 32 rows per wave => 8 f32 accumulators, 16 WMMAs per tap (432 total).
// ---------------------------------------------------------------------------
__global__ void __launch_bounds__(256, 1)
conv27_kernel(const unsigned short* __restrict__ Xb,
              const unsigned* __restrict__ Wpk,     // 27 packed mats
              const int* __restrict__ nbr,          // [M,27]
              const float* __restrict__ scale,
              const float* __restrict__ bias,
              const unsigned short* __restrict__ zrow,
              unsigned short* __restrict__ outb,
              int mrows) {
    extern __shared__ U4 ldsWq[];                // 27*512 U4 = 216 KB
    int tid = threadIdx.x;

    // ---- stage all 27 B-fragment matrices into LDS via async DMA path -----
    {
        unsigned ldsbase = (unsigned)(size_t)(void*)ldsWq;
        for (int c = tid; c < TAPS * 512; c += 256) {   // 16B chunks
            unsigned loff = ldsbase + (unsigned)c * 16u;
            const unsigned* g = Wpk + (size_t)c * 4;
            asm volatile("global_load_async_to_lds_b128 %0, %1, off"
                         :: "v"(loff), "v"(g) : "memory");
        }
        asm volatile("s_wait_asynccnt 0x0" ::: "memory");
    }
    __syncthreads();

    int wave = tid >> 5, ln = tid & 31;
    int rowBase = blockIdx.x * 256 + wave * 32;  // two 16-row tiles per wave
    if (rowBase >= mrows) return;
    int lr = ln & 15, hs = ln >> 4;
    const int* nrow0 = nbr + (size_t)(rowBase + lr) * TAPS;
    const int* nrow1 = nbr + (size_t)(rowBase + 16 + lr) * TAPS;

    v8f acc[8];
#pragma unroll
    for (int i = 0; i < 8; ++i) acc[i] = (v8f)0.0f;

    // branchless masked gather: invalid neighbor -> shared zero row
    auto rowptr = [&](int idx) -> const unsigned short* {
        return idx >= 0 ? Xb + (size_t)idx * CCH : zrow;
    };

    const unsigned short* r0 = rowptr(nrow0[0]);
    const unsigned short* r1 = rowptr(nrow1[0]);
    v16bf a00 = load_a(r0, 0, hs), a01 = load_a(r0, 32, hs);
    v16bf a10 = load_a(r1, 0, hs), a11 = load_a(r1, 32, hs);

    for (int k = 0; k < TAPS; ++k) {
        // issue next tap's gathers before this tap's math (software pipeline)
        int kn = (k + 1 < TAPS) ? k + 1 : k;
        const unsigned short* n0 = rowptr(nrow0[kn]);
        const unsigned short* n1 = rowptr(nrow1[kn]);
        v16bf na00 = load_a(n0, 0, hs), na01 = load_a(n0, 32, hs);
        v16bf na10 = load_a(n1, 0, hs), na11 = load_a(n1, 32, hs);
        if (k + 2 < TAPS) {
            int p0 = nrow0[k + 2];
            if (p0 >= 0) __builtin_prefetch(Xb + (size_t)p0 * CCH, 0, 3);
        }

        int kb4 = k * 512;                        // U4 index of tap k
#pragma unroll
        for (int kc = 0; kc < 2; ++kc) {
            v16bf acur0 = kc ? a01 : a00;
            v16bf acur1 = kc ? a11 : a10;
#pragma unroll
            for (int nt = 0; nt < 4; ++nt) {
                Frag bf;
                int bi4 = kb4 + ((kc * 4 + nt) * 32 + ln) * 2;
                bf.q[0] = ldsWq[bi4];             // ds_load_b128
                bf.q[1] = ldsWq[bi4 + 1];         // ds_load_b128
                acc[nt]     = wmma_bf16(acur0, bf.v, acc[nt]);
                acc[4 + nt] = wmma_bf16(acur1, bf.v, acc[4 + nt]);
            }
        }
        a00 = na00; a01 = na01; a10 = na10; a11 = na11;
    }

#pragma unroll
    for (int t = 0; t < 2; ++t) {
        int rb = rowBase + t * 16;
#pragma unroll
        for (int nt = 0; nt < 4; ++nt) {
            int col = nt * 16 + lr;
            float sc = scale[col], bb = bias[col];
#pragma unroll
            for (int j = 0; j < 8; ++j) {
                int row = rb + hs * 8 + j;
                float y = acc[t * 4 + nt][j] * sc + bb;
                y = y > 0.f ? y : 0.01f * y;
                outb[(size_t)row * CCH + col] = f2bf(y);
            }
        }
    }
}

// V = LeakyReLU(h + feats, 0.1) in fp32
__global__ void resid_kernel(const unsigned short* __restrict__ hb,
                             const float* __restrict__ feats,
                             float* __restrict__ V, int n) {
    int t = blockIdx.x * blockDim.x + threadIdx.x;
    if (t >= n) return;
    float x = bf2f(hb[t]) + feats[t];
    V[t] = x > 0.f ? x : 0.1f * x;
}

// per-scale scatter: ssum[seg,c] += V[m,c]; cnt[seg] += 1 (c==0 lane)
__global__ void scatter_kernel(const float* __restrict__ V,
                               const int* __restrict__ inv,
                               float* __restrict__ ssum,
                               float* __restrict__ cnt, int n) {
    int t = blockIdx.x * blockDim.x + threadIdx.x;
    if (t >= n) return;
    int m = t >> 6, c = t & 63;
    int seg = inv[m];
    atomicAdd(&ssum[(size_t)seg * CCH + c], V[t]);
    if (c == 0) atomicAdd(&cnt[seg], 1.0f);
}

// Os = (V - mean[seg]) * V  -> bf16
__global__ void os_kernel(const float* __restrict__ V,
                          const int* __restrict__ inv,
                          const float* __restrict__ ssum,
                          const float* __restrict__ cnt,
                          unsigned short* __restrict__ osb, int n) {
    int t = blockIdx.x * blockDim.x + threadIdx.x;
    if (t >= n) return;
    int m = t >> 6, c = t & 63;
    int seg = inv[m];
    float cn = cnt[seg]; cn = cn > 1.f ? cn : 1.f;
    float mean = ssum[(size_t)seg * CCH + c] / cn;
    float v = V[t];
    osb[t] = f2bf((v - mean) * v);
}

// sum_x = sum_i ms[i] (bf16 in, bf16 out)
__global__ void sumx_kernel(const unsigned short* __restrict__ ms0,
                            const unsigned short* __restrict__ ms1,
                            const unsigned short* __restrict__ ms2,
                            const unsigned short* __restrict__ ms3,
                            unsigned short* __restrict__ out, int n) {
    int t = blockIdx.x * blockDim.x + threadIdx.x;
    if (t >= n) return;
    float s = bf2f(ms0[t]) + bf2f(ms1[t]) + bf2f(ms2[t]) + bf2f(ms3[t]);
    out[t] = f2bf(s);
}

// heads: out[M,N] = X[M,64] @ W[64,N] + b
__global__ void head_kernel(const float* __restrict__ X,
                            const float* __restrict__ W,
                            const float* __restrict__ b,
                            float* __restrict__ out, int mrows, int N) {
    int t = blockIdx.x * blockDim.x + threadIdx.x;
    if (t >= mrows * N) return;
    int m = t / N, n = t % N;
    const float* xr = X + (size_t)m * CCH;
    float s = b[n];
#pragma unroll
    for (int c = 0; c < CCH; ++c) s += xr[c] * W[c * N + n];
    out[t] = s;
}

// ---------------------------------------------------------------------------
extern "C" void kernel_launch(void* const* d_in, const int* in_sizes, int n_in,
                              void* d_out, int out_size, void* d_ws, size_t ws_size,
                              hipStream_t stream) {
    (void)in_sizes; (void)n_in; (void)out_size; (void)ws_size;
    const float* feats   = (const float*)d_in[0];
    const float* w1a     = (const float*)d_in[1];
    const float* s1a     = (const float*)d_in[2];
    const float* b1a     = (const float*)d_in[3];
    const float* w3      = (const float*)d_in[4];
    const float* s3      = (const float*)d_in[5];
    const float* b3      = (const float*)d_in[6];
    const float* w1b     = (const float*)d_in[7];
    const float* s1b     = (const float*)d_in[8];
    const float* b1b     = (const float*)d_in[9];
    const float* aux_w   = (const float*)d_in[10];
    const float* aux_b   = (const float*)d_in[11];
    const float* proj_w  = (const float*)d_in[12];
    const float* proj_b  = (const float*)d_in[13];
    const float* proj_s  = (const float*)d_in[14];
    const float* proj_bb = (const float*)d_in[15];
    const float* attn_w  = (const float*)d_in[16];
    const float* attn_b  = (const float*)d_in[17];
    const float* attn_s  = (const float*)d_in[18];
    const float* attn_bb = (const float*)d_in[19];
    const float* head_w  = (const float*)d_in[20];
    const float* head_b  = (const float*)d_in[21];
    const int*   nbr     = (const int*)d_in[22];
    const int*   inv     = (const int*)d_in[23];

    const int M  = M_PTS;
    const int MC = M * CCH;

    // ---- workspace layout --------------------------------------------------
    char* ws = (char*)d_ws;
    size_t off = 0;
    auto take = [&](size_t bytes) { char* p = ws + off; off = (off + bytes + 255) & ~(size_t)255; return p; };
    unsigned* pk_w1a  = (unsigned*)take((size_t)4   * 2048 * 4);
    unsigned* pk_w3   = (unsigned*)take((size_t)108 * 2048 * 4);
    unsigned* pk_w1b  = (unsigned*)take((size_t)4   * 2048 * 4);
    unsigned* pk_proj = (unsigned*)take((size_t)4   * 2048 * 4);
    unsigned* pk_attn = (unsigned*)take((size_t)4   * 2048 * 4);
    unsigned short* zrow = (unsigned short*)take(128);           // 64 bf16 zeros
    unsigned short* hb = (unsigned short*)take((size_t)MC * 2);
    unsigned short* t1 = (unsigned short*)take((size_t)MC * 2);  // also Os buffer
    unsigned short* t2 = (unsigned short*)take((size_t)MC * 2);  // also sum_x
    float* Vf   = (float*)take((size_t)MC * 4);
    float* ssum = (float*)take((size_t)MC * 4);
    float* cnt  = (float*)take((size_t)M * 4);
    unsigned short* ms[4];
    for (int i = 0; i < 4; ++i) ms[i] = (unsigned short*)take((size_t)MC * 2);

    float* fused  = (float*)d_out;             // [M,64]
    float* logits = fused + (size_t)MC;        // [M,20]
    float* auxout = logits + (size_t)M * NCLS; // [M,20]

    const int EB = 256;
    int ew_grid    = (MC + EB - 1) / EB;
    int gemm_grid  = (M + 127) / 128;          // 8 waves x 16 rows
    int conv_grid  = (M + 255) / 256;          // 8 waves x 32 rows
    size_t convLds = (size_t)TAPS * 2048 * 4;  // 216 KB dynamic LDS

    // ---- weight pre-swizzle to WMMA B-fragment layout ----------------------
    auto pack = [&](const float* src, unsigned* dst, int nmat) {
        int n = nmat * 2048;
        hipLaunchKernelGGL(pack_w_kernel, dim3((n + EB - 1) / EB), dim3(EB), 0, stream, src, dst, nmat);
    };
    pack(w1a, pk_w1a, 4);
    pack(w3,  pk_w3, 108);
    pack(w1b, pk_w1b, 4);
    pack(proj_w, pk_proj, 4);
    pack(attn_w, pk_attn, 4);

    hipMemsetAsync(zrow, 0, 128, stream);
    hipLaunchKernelGGL(cvt_bf16_kernel, dim3(ew_grid), dim3(EB), 0, stream, feats, hb, MC);

    // ---- SFE block: 4 x (1x1 -> 3x3x3 submanifold -> 1x1) ------------------
    for (int i = 0; i < 4; ++i) {
        hipLaunchKernelGGL((gemm64_kernel<0>), dim3(gemm_grid), dim3(EB), 0, stream,
                           hb, pk_w1a + (size_t)i * 2048, (const float*)nullptr,
                           s1a + i * CCH, b1a + i * CCH, t1, nullptr, nullptr, 0, M);
        hipLaunchKernelGGL(conv27_kernel, dim3(conv_grid), dim3(EB), convLds, stream,
                           t1, pk_w3 + (size_t)i * TAPS * 2048, nbr,
                           s3 + i * CCH, b3 + i * CCH, zrow, t2, M);
        hipLaunchKernelGGL((gemm64_kernel<0>), dim3(gemm_grid), dim3(EB), 0, stream,
                           t2, pk_w1b + (size_t)i * 2048, (const float*)nullptr,
                           s1b + i * CCH, b1b + i * CCH, hb, nullptr, nullptr, 0, M);
    }

    // ---- residual + aux head ----------------------------------------------
    hipLaunchKernelGGL(resid_kernel, dim3(ew_grid), dim3(EB), 0, stream, hb, feats, Vf, MC);
    hipLaunchKernelGGL(head_kernel, dim3((M * NCLS + EB - 1) / EB), dim3(EB), 0, stream,
                       Vf, aux_w, aux_b, auxout, M, NCLS);

    // ---- multi-scale projection -------------------------------------------
    for (int i = 0; i < 4; ++i) {
        hipMemsetAsync(ssum, 0, (size_t)MC * 4, stream);
        hipMemsetAsync(cnt,  0, (size_t)M * 4, stream);
        hipLaunchKernelGGL(scatter_kernel, dim3(ew_grid), dim3(EB), 0, stream,
                           Vf, inv + (size_t)i * M, ssum, cnt, MC);
        hipLaunchKernelGGL(os_kernel, dim3(ew_grid), dim3(EB), 0, stream,
                           Vf, inv + (size_t)i * M, ssum, cnt, t1, MC);
        hipLaunchKernelGGL((gemm64_kernel<0>), dim3(gemm_grid), dim3(EB), 0, stream,
                           t1, pk_proj + (size_t)i * 2048, proj_b + i * CCH,
                           proj_s + i * CCH, proj_bb + i * CCH, ms[i], nullptr, nullptr, 0, M);
    }

    // ---- attentive fusion ---------------------------------------------------
    hipLaunchKernelGGL(sumx_kernel, dim3(ew_grid), dim3(EB), 0, stream,
                       ms[0], ms[1], ms[2], ms[3], t2, MC);
    for (int i = 0; i < 4; ++i) {
        hipLaunchKernelGGL((gemm64_kernel<1>), dim3(gemm_grid), dim3(EB), 0, stream,
                           t2, pk_attn + (size_t)i * 2048, attn_b + i * CCH,
                           attn_s + i * CCH, attn_bb + i * CCH,
                           (unsigned short*)nullptr, ms[i], fused, (i > 0) ? 1 : 0, M);
    }

    // ---- classifier head ----------------------------------------------------
    hipLaunchKernelGGL(head_kernel, dim3((M * NCLS + EB - 1) / EB), dim3(EB), 0, stream,
                       fused, head_w, head_b, logits, M, NCLS);
}
